// InvariantPointAttention_48928267436132
// MI455X (gfx1250) — matrix operand order repaired and checked by reading, not compiled, not run on hardware
//
#include <hip/hip_runtime.h>
#include <hip/hip_bf16.h>
#include <math.h>

#define DEVFN __device__ __forceinline__

typedef __attribute__((ext_vector_type(16))) __bf16 v16bf;
typedef __attribute__((ext_vector_type(8)))  float  v8f;
typedef __attribute__((ext_vector_type(8)))  unsigned int u32x8;
typedef __attribute__((ext_vector_type(4)))  unsigned int u32x4;
typedef __attribute__((ext_vector_type(8)))  int i32x8;
typedef __attribute__((ext_vector_type(4)))  int i32x4;

#if defined(__has_builtin)
#if __has_builtin(__builtin_amdgcn_tensor_load_to_lds) && __has_builtin(__builtin_amdgcn_s_wait_tensorcnt)
#define IPA_HAS_TDM 1
#endif
#endif
#ifndef IPA_HAS_TDM
#define IPA_HAS_TDM 0
#endif

namespace ipa {
constexpr int    N     = 1024;
constexpr int    CS    = 384;
constexpr int    CZ    = 128;
constexpr int    CH    = 16;
constexpr int    H     = 12;
constexpr int    PQK   = 4;
constexpr int    PV    = 8;
constexpr size_t NN    = (size_t)N * (size_t)N;
constexpr int    HCH   = H * CH;                 // 192
constexpr int    H2CH  = 2 * H * CH;             // 384
constexpr int    HQP3  = H * PQK * 3;            // 144
constexpr int    HKVP3 = H * (PQK + PV) * 3;     // 432
constexpr int    DCAT  = H * (CZ + CH + PV * 4); // 2112
constexpr float  INFV  = 100000.0f;
constexpr float  EPSV  = 1e-8f;
constexpr float  SQK   = 0.14433756729740643f;   // sqrt(1/48)
constexpr float  HWS   = 0.13608276348795434f;   // sqrt(1/54)
constexpr float  SB3   = 0.5773502691896258f;    // sqrt(1/3)
} // namespace ipa

// ---------------------------------------------------------------------------
// bf16 packing + WMMA fragment helpers (CDNA5 VGPR layouts)
// ---------------------------------------------------------------------------
DEVFN __bf16 to_bf16(float f) {  // scalar RNE (used in small kernels only)
  unsigned u = __builtin_bit_cast(unsigned, f);
  unsigned r = (u + 0x7FFFu + ((u >> 16) & 1u)) >> 16;
  unsigned short h = (unsigned short)r;
  return __builtin_bit_cast(__bf16, h);
}
// two f32 -> packed bf16 pair (lo in [15:0]) via one v_perm_b32
DEVFN unsigned pack_bf16(float flo, float fhi) {
  unsigned a = __builtin_bit_cast(unsigned, flo) + 0x8000u;  // round-to-nearest
  unsigned b = __builtin_bit_cast(unsigned, fhi) + 0x8000u;
#if defined(__has_builtin) && __has_builtin(__builtin_amdgcn_perm)
  return __builtin_amdgcn_perm(b, a, 0x07060302u);  // {b[31:16], a[31:16]}
#else
  return (b & 0xFFFF0000u) | (a >> 16);
#endif
}

// A 16x32: lanes 0-15: M=lane, K in {0..15}; lanes 16-31: M=lane-16, K in {16..31}
DEVFN v16bf frag_a_f32(const float* sa, int ld) {
  const int lane = threadIdx.x & 31;
  const int row = lane & 15, hi = lane >> 4;
  u32x8 u;
#pragma unroll
  for (int p = 0; p < 8; ++p) {
    int kb = 2 * p + 16 * hi;
    u[p] = pack_bf16(sa[row * ld + kb], sa[row * ld + kb + 1]);
  }
  return __builtin_bit_cast(v16bf, u);
}
DEVFN v16bf frag_a_bf16(const __bf16* sa) {  // sa row-major [16][32], pairs are u32
  const int lane = threadIdx.x & 31;
  const int row = lane & 15, hi = lane >> 4;
  const unsigned* sp = (const unsigned*)sa;
  u32x8 u;
#pragma unroll
  for (int p = 0; p < 8; ++p) {
    int kb = 2 * p + 16 * hi;
    u[p] = sp[(row * 32 + kb) >> 1];
  }
  return __builtin_bit_cast(v16bf, u);
}
// B 32x16: lanes 0-15 hold K=0..15 (VGPR i -> K=2i,2i+1); lanes 16-31 K=16..31
DEVFN v16bf frag_b_f32(const float* sb, int ld) {  // sb row-major [32][cols]
  const int lane = threadIdx.x & 31;
  const int col = lane & 15, koff = (lane >> 4) * 16;
  u32x8 u;
#pragma unroll
  for (int i = 0; i < 8; ++i)
    u[i] = pack_bf16(sb[(koff + 2 * i) * ld + col], sb[(koff + 2 * i + 1) * ld + col]);
  return __builtin_bit_cast(v16bf, u);
}
DEVFN v16bf frag_b_bf16T(const __bf16* sbT) {  // col-major [16 cols][32 k]
  const int lane = threadIdx.x & 31;
  const int col = lane & 15, koff = (lane >> 4) * 16;
  const unsigned* sp = (const unsigned*)(sbT + col * 32 + koff);
  u32x8 u;
#pragma unroll
  for (int i = 0; i < 8; ++i) u[i] = sp[i];  // 8 consecutive u32 -> b128 loads
  return __builtin_bit_cast(v16bf, u);
}

DEVFN v8f wmma_bf16(v16bf a, v16bf b, v8f c) {
  return __builtin_amdgcn_wmma_f32_16x16x32_bf16(false, a, false, b, (short)0, c,
                                                 false, false);
}
DEVFN v8f zero8() { v8f z = {0.f,0.f,0.f,0.f,0.f,0.f,0.f,0.f}; return z; }
// C/D 16x16 f32: lanes 0-15: N=lane, M=v; lanes 16-31: N=lane-16, M=v+8

#if IPA_HAS_TDM
// Issue one TDM 2D-tile load: rows x cols fp32, row stride (all in f32 units).
DEVFN void tdm_load_f32(const float* gsrc, void* lds_dst, int cols, int rows,
                        int row_stride) {
  unsigned long long ga = (unsigned long long)gsrc;
  unsigned lds_off = (unsigned)(unsigned long long)lds_dst;  // addr[31:0] = LDS offset
  u32x4 g0 = {0u, 0u, 0u, 0u};
  g0[0] = 1u;                                                 // count=1
  g0[1] = lds_off;                                            // lds_addr [63:32]
  g0[2] = (unsigned)ga;                                       // global_addr [95:64]
  g0[3] = (unsigned)((ga >> 32) & 0x01FFFFFFull) | (2u << 30);// addr[120:96] | type=2
  i32x8 g1 = {0, 0, 0, 0, 0, 0, 0, 0};
  g1[0] = (2 << 16);                    // data_size = 4 bytes
  g1[1] = (cols << 16);                 // tensor_dim0 low16
  g1[2] = ((unsigned)cols >> 16) | ((unsigned)rows << 16);  // dim0 hi | dim1 lo
  g1[3] = ((unsigned)rows >> 16) | ((unsigned)cols << 16);  // dim1 hi | tile_dim0
  g1[4] = rows;                         // tile_dim1 (tile_dim2 = 0)
  g1[5] = row_stride;                   // tensor_dim0_stride low32
  i32x4 gz4 = {0, 0, 0, 0};
#if __has_include(<hip/amd_detail/amd_gfx1250_TDM.h>)
  i32x8 gz8 = {0, 0, 0, 0, 0, 0, 0, 0};
  __builtin_amdgcn_tensor_load_to_lds(g0, g1, gz4, gz4, gz8, 0);
#else
  __builtin_amdgcn_tensor_load_to_lds(g0, g1, gz4, gz4, 0);
#endif
}
#endif

// ---------------------------------------------------------------------------
// Generic C[MxP] = A[MxK] @ W[KxP] (+bias), 64-row M tile: 4 WMMA per K-step.
// ---------------------------------------------------------------------------
__global__ __launch_bounds__(32)
void k_gemm(const float* __restrict__ A, int lda,
            const float* __restrict__ W, int ldw,
            const float* __restrict__ bias,
            float* __restrict__ C, int ldc, int K, int P, float scale) {
  __shared__ float sa[64 * 32];
  __shared__ float sb[32 * 16];
  const int lane = threadIdx.x;
  const int i0 = blockIdx.x * 64, j0 = blockIdx.y * 16;
  v8f acc[4];
#pragma unroll
  for (int m = 0; m < 4; ++m) acc[m] = zero8();
  for (int k0 = 0; k0 < K; k0 += 32) {
    for (int e = lane; e < 64 * 32; e += 32) {
      int r = e >> 5, c = e & 31;
      sa[e] = (k0 + c < K) ? A[(size_t)(i0 + r) * lda + k0 + c] : 0.f;
    }
    for (int e = lane; e < 32 * 16; e += 32) {
      int r = e >> 4, c = e & 15;
      sb[e] = (k0 + r < K && j0 + c < P) ? W[(size_t)(k0 + r) * ldw + j0 + c] : 0.f;
    }
    if (k0 + 32 < K)
      __builtin_prefetch(&A[(size_t)(i0 + lane) * lda + k0 + 32], 0, 3);
    __syncthreads();
    v16bf bf = frag_b_f32(sb, 16);
#pragma unroll
    for (int m = 0; m < 4; ++m)
      acc[m] = wmma_bf16(frag_a_f32(sa + m * 16 * 32, 32), bf, acc[m]);
    __syncthreads();
  }
  const int n = lane & 15, hi = lane >> 4;
  if (j0 + n < P) {
    float bv = bias ? bias[j0 + n] : 0.f;
#pragma unroll
    for (int m = 0; m < 4; ++m)
#pragma unroll
      for (int v = 0; v < 8; ++v)
        C[(size_t)(i0 + m * 16 + v + 8 * hi) * ldc + j0 + n] = scale * (acc[m][v] + bv);
  }
}

// ---------------------------------------------------------------------------
// bias[h,i,j] = sqrt(1/3)*(z[i,j,:] @ wb[:,h] + bb[h])   (z pass 1)
// 64x128 fp32 z tile fetched by one tensor_load_to_lds, synced via TENSORcnt.
// ---------------------------------------------------------------------------
__global__ __launch_bounds__(32)
void k_bias(const float* __restrict__ z, const float* __restrict__ wb,
            const float* __restrict__ bb, float* __restrict__ Abuf) {
  using namespace ipa;
  __shared__ float zl[64 * CZ];     // 32 KB
  __shared__ float sbw[CZ * 16];    // 8 KB
  const int lane = threadIdx.x;
  const size_t row0 = (size_t)blockIdx.x * 64;  // flattened i*N+j

  for (int e = lane; e < CZ * 16; e += 32) {
    int r = e >> 4, c = e & 15;
    sbw[e] = (c < H) ? wb[r * H + c] : 0.f;
  }
#if IPA_HAS_TDM
  tdm_load_f32(z + row0 * CZ, zl, CZ, 64, CZ);
  __builtin_amdgcn_s_wait_tensorcnt(0);
#else
  for (int e = lane; e < (64 * CZ) / 4; e += 32)
    ((float4*)zl)[e] = ((const float4*)(z + row0 * CZ))[e];
#endif
  __syncthreads();

  v8f acc[4];
#pragma unroll
  for (int m = 0; m < 4; ++m) acc[m] = zero8();
#pragma unroll
  for (int k0 = 0; k0 < CZ; k0 += 32) {
    v16bf bf = frag_b_f32(sbw + k0 * 16, 16);
#pragma unroll
    for (int m = 0; m < 4; ++m)
      acc[m] = wmma_bf16(frag_a_f32(zl + m * 16 * CZ + k0, CZ), bf, acc[m]);
  }
  const int n = lane & 15, hi = lane >> 4;
  if (n < H) {
#pragma unroll
    for (int m = 0; m < 4; ++m)
#pragma unroll
      for (int v = 0; v < 8; ++v)
        Abuf[(size_t)n * NN + row0 + m * 16 + v + 8 * hi] = SB3 * (acc[m][v] + bb[n]);
  }
}

// ---------------------------------------------------------------------------
// Per-residue frames, point norms, fused bf16 q/k vectors
// ---------------------------------------------------------------------------
__global__ __launch_bounds__(384)
void k_points(const float* __restrict__ qraw, const float* __restrict__ kvraw,
              const float* __restrict__ qpraw, const float* __restrict__ kvpraw,
              const float* __restrict__ rot, const float* __restrict__ trans,
              const float* __restrict__ head_w,
              float* __restrict__ qn, float* __restrict__ kn,
              float* __restrict__ vpts, __bf16* __restrict__ qcat,
              __bf16* __restrict__ kcat) {
  using namespace ipa;
  __shared__ float s_qp[48 * 3];
  __shared__ float s_kp[48 * 3];
  __shared__ float s_qn[H];
  __shared__ float s_kn[H];
  const int n = blockIdx.x;
  const int tid = threadIdx.x;
  if (tid < H) s_qn[tid] = 0.f;
  else if (tid < 2 * H) s_kn[tid - H] = 0.f;
  __syncthreads();

  const float* R = rot + (size_t)n * 9;
  const float* T = trans + (size_t)n * 3;
  if (tid < 48) {                 // q points (pf = h*4+p), raw = [x|y|z] blocks
    int pf = tid;
    float x0 = qpraw[(size_t)n * HQP3 + pf];
    float y0 = qpraw[(size_t)n * HQP3 + 48 + pf];
    float z0 = qpraw[(size_t)n * HQP3 + 96 + pf];
    float sq = 0.f;
#pragma unroll
    for (int x = 0; x < 3; ++x) {
      float o = R[x * 3] * x0 + R[x * 3 + 1] * y0 + R[x * 3 + 2] * z0 + T[x];
      s_qp[pf * 3 + x] = o;
      sq += o * o;
    }
    atomicAdd(&s_qn[pf >> 2], sq);
  } else if (tid < 96) {          // k points
    int pf = tid - 48, h = pf >> 2, p = pf & 3, pf2 = h * 12 + p;
    float x0 = kvpraw[(size_t)n * HKVP3 + pf2];
    float y0 = kvpraw[(size_t)n * HKVP3 + 144 + pf2];
    float z0 = kvpraw[(size_t)n * HKVP3 + 288 + pf2];
    float sq = 0.f;
#pragma unroll
    for (int x = 0; x < 3; ++x) {
      float o = R[x * 3] * x0 + R[x * 3 + 1] * y0 + R[x * 3 + 2] * z0 + T[x];
      s_kp[pf * 3 + x] = o;
      sq += o * o;
    }
    atomicAdd(&s_kn[h], sq);
  } else if (tid < 192) {         // v points, stored padded [N][H][32]
    int vi = tid - 96, h = vi >> 3, p = vi & 7, pf2 = h * 12 + 4 + p;
    float x0 = kvpraw[(size_t)n * HKVP3 + pf2];
    float y0 = kvpraw[(size_t)n * HKVP3 + 144 + pf2];
    float z0 = kvpraw[(size_t)n * HKVP3 + 288 + pf2];
#pragma unroll
    for (int x = 0; x < 3; ++x) {
      float o = R[x * 3] * x0 + R[x * 3 + 1] * y0 + R[x * 3 + 2] * z0 + T[x];
      vpts[((size_t)n * H + h) * 32 + p * 3 + x] = o;
    }
  } else if (tid < 288) {         // zero pad slots 24..31
    int id = tid - 192, h = id >> 3, slot = 24 + (id & 7);
    vpts[((size_t)n * H + h) * 32 + slot] = 0.f;
  }
  __syncthreads();

  const int h = tid >> 5, slot = tid & 31;  // 384 = 12 heads x 32 slots
  const float hwv = log1pf(expf(head_w[h])) * HWS;
  __bf16 qv = (__bf16)0.0f, kv = (__bf16)0.0f;
  if (slot < 16) {
    qv = to_bf16(qraw[(size_t)n * HCH + h * CH + slot] * SQK);
    kv = to_bf16(kvraw[(size_t)n * H2CH + h * 32 + slot]);
  } else if (slot < 28) {
    int d = slot - 16;
    qv = to_bf16(s_qp[h * 12 + d] * hwv);
    kv = to_bf16(s_kp[h * 12 + d]);
  }
  qcat[((size_t)h * N + n) * 32 + slot] = qv;
  kcat[((size_t)h * N + n) * 32 + slot] = kv;
  if (tid < H) qn[(size_t)n * H + tid] = s_qn[tid];
  else if (tid < 2 * H) kn[(size_t)n * H + tid - H] = s_kn[tid - H];
}

// ---------------------------------------------------------------------------
// logits: A[h,i,j] += qcat_i . kcat_j - 0.5*hw*(qn_i+kn_j) + mask
// A fragments built once per block (bulk b128 copies, zero conversion work).
// ---------------------------------------------------------------------------
__global__ __launch_bounds__(32)
void k_logits(const __bf16* __restrict__ qcat, const __bf16* __restrict__ kcat,
              const float* __restrict__ qn, const float* __restrict__ kn,
              const float* __restrict__ mask, const float* __restrict__ head_w,
              float* __restrict__ Abuf) {
  using namespace ipa;
  __shared__ __bf16 sa[64 * 32];   // row-major qcat tile
  __shared__ __bf16 sbT[16 * 32];  // kcat tile, [col][k] = source row-major layout
  const int lane = threadIdx.x;
  const int i0 = blockIdx.x * 64, h = blockIdx.y;
  const float hwv = log1pf(expf(head_w[h])) * HWS;
  {
    const uint4* src = (const uint4*)(qcat + ((size_t)h * N + i0) * 32);
    for (int e = lane; e < (64 * 32 * 2) / 16; e += 32) ((uint4*)sa)[e] = src[e];
  }
  __syncthreads();
  v16bf af[4];
#pragma unroll
  for (int m = 0; m < 4; ++m) af[m] = frag_a_bf16(sa + m * 16 * 32);
  const int n = lane & 15, hi = lane >> 4;

  for (int j0 = 0; j0 < N; j0 += 16) {
    const uint4* src = (const uint4*)(kcat + ((size_t)h * N + j0) * 32);
    for (int e = lane; e < (16 * 32 * 2) / 16; e += 32) ((uint4*)sbT)[e] = src[e];
    __syncthreads();
    v16bf bf = frag_b_bf16T(sbT);
    const int j = j0 + n;
    const float knj = kn[(size_t)j * H + h];
    const float mj = mask[j];
#pragma unroll
    for (int m = 0; m < 4; ++m) {
      v8f acc = zero8();
      acc = wmma_bf16(af[m], bf, acc);
#pragma unroll
      for (int v = 0; v < 8; ++v) {
        int i = i0 + m * 16 + v + 8 * hi;
        size_t off = (size_t)h * NN + (size_t)i * N + j;
        Abuf[off] = acc[v] + Abuf[off]
                  - 0.5f * hwv * (qn[(size_t)i * H + h] + knj)
                  + INFV * (mask[i] * mj - 1.f);
      }
    }
    __syncthreads();
  }
}

__global__ __launch_bounds__(256)
void k_softmax(float* __restrict__ Abuf) {
  using namespace ipa;
  __shared__ float red[256];
  const int i = blockIdx.x, h = blockIdx.y, tid = threadIdx.x;
  float* row = Abuf + (size_t)h * NN + (size_t)i * N;
  float v[4], mx = -3.4e38f;
#pragma unroll
  for (int t = 0; t < 4; ++t) { v[t] = row[tid + t * 256]; mx = fmaxf(mx, v[t]); }
  red[tid] = mx; __syncthreads();
  for (int s = 128; s > 0; s >>= 1) { if (tid < s) red[tid] = fmaxf(red[tid], red[tid + s]); __syncthreads(); }
  mx = red[0]; __syncthreads();
  float sum = 0.f;
#pragma unroll
  for (int t = 0; t < 4; ++t) { v[t] = __expf(v[t] - mx); sum += v[t]; }
  red[tid] = sum; __syncthreads();
  for (int s = 128; s > 0; s >>= 1) { if (tid < s) red[tid] += red[tid + s]; __syncthreads(); }
  const float inv = 1.f / red[0];
#pragma unroll
  for (int t = 0; t < 4; ++t) row[tid + t * 256] = v[t] * inv;
}

// o[i, h*16+c] = sum_j a[h,i,j] v[j,h,c]
__global__ __launch_bounds__(32)
void k_out_o(const float* __restrict__ Abuf, const float* __restrict__ kvraw,
             float* __restrict__ cat) {
  using namespace ipa;
  __shared__ float sa[64 * 32];
  __shared__ float sb[32 * 16];
  const int lane = threadIdx.x;
  const int i0 = blockIdx.x * 64, h = blockIdx.y;
  v8f acc[4];
#pragma unroll
  for (int m = 0; m < 4; ++m) acc[m] = zero8();
  for (int k0 = 0; k0 < N; k0 += 32) {
    for (int e = lane; e < 64 * 32; e += 32) {
      int r = e >> 5, c = e & 31;
      sa[e] = Abuf[(size_t)h * NN + (size_t)(i0 + r) * N + k0 + c];
    }
    for (int e = lane; e < 512; e += 32) {
      int r = e >> 4, c = e & 15;
      sb[e] = kvraw[(size_t)(k0 + r) * H2CH + h * 32 + 16 + c];
    }
    __syncthreads();
    v16bf bf = frag_b_f32(sb, 16);
#pragma unroll
    for (int m = 0; m < 4; ++m)
      acc[m] = wmma_bf16(frag_a_f32(sa + m * 16 * 32, 32), bf, acc[m]);
    __syncthreads();
  }
  const int n = lane & 15, hi = lane >> 4;
#pragma unroll
  for (int m = 0; m < 4; ++m)
#pragma unroll
    for (int v = 0; v < 8; ++v)
      cat[(size_t)(i0 + m * 16 + v + 8 * hi) * DCAT + h * CH + n] = acc[m][v];
}

// o_pt raw: optraw[i][h*32+s] = sum_j a[h,i,j] vpts[j][h*32+s]
__global__ __launch_bounds__(32)
void k_out_opt(const float* __restrict__ Abuf, const float* __restrict__ vpts,
               float* __restrict__ optraw) {
  using namespace ipa;
  __shared__ float sa[64 * 32];
  __shared__ float sb[32 * 32];
  const int lane = threadIdx.x;
  const int i0 = blockIdx.x * 64, h = blockIdx.y;
  v8f acc[4][2];
#pragma unroll
  for (int m = 0; m < 4; ++m) { acc[m][0] = zero8(); acc[m][1] = zero8(); }
  for (int k0 = 0; k0 < N; k0 += 32) {
    for (int e = lane; e < 64 * 32; e += 32) {
      int r = e >> 5, c = e & 31;
      sa[e] = Abuf[(size_t)h * NN + (size_t)(i0 + r) * N + k0 + c];
    }
    for (int e = lane; e < 32 * 32; e += 32) {
      int r = e >> 5, c = e & 31;
      sb[e] = vpts[((size_t)(k0 + r) * H + h) * 32 + c];
    }
    __syncthreads();
    v16bf bf0 = frag_b_f32(sb, 32);
    v16bf bf1 = frag_b_f32(sb + 16, 32);
#pragma unroll
    for (int m = 0; m < 4; ++m) {
      v16bf afm = frag_a_f32(sa + m * 16 * 32, 32);
      acc[m][0] = wmma_bf16(afm, bf0, acc[m][0]);
      acc[m][1] = wmma_bf16(afm, bf1, acc[m][1]);
    }
    __syncthreads();
  }
  const int n = lane & 15, hi = lane >> 4;
#pragma unroll
  for (int m = 0; m < 4; ++m)
#pragma unroll
    for (int ct = 0; ct < 2; ++ct)
#pragma unroll
      for (int v = 0; v < 8; ++v)
        optraw[((size_t)(i0 + m * 16 + v + 8 * hi) * H + h) * 32 + ct * 16 + n] =
            acc[m][ct][v];
}

// inverse frame + norms -> cat cols [192,576)
__global__ __launch_bounds__(96)
void k_opt_finish(const float* __restrict__ optraw, const float* __restrict__ rot,
                  const float* __restrict__ trans, float* __restrict__ cat) {
  using namespace ipa;
  const int n = blockIdx.x, tid = threadIdx.x;
  const int h = tid >> 3, p = tid & 7;
  const float* R = rot + (size_t)n * 9;
  const float* T = trans + (size_t)n * 3;
  float d0 = optraw[((size_t)n * H + h) * 32 + p * 3 + 0] - T[0];
  float d1 = optraw[((size_t)n * H + h) * 32 + p * 3 + 1] - T[1];
  float d2 = optraw[((size_t)n * H + h) * 32 + p * 3 + 2] - T[2];
  float l[3];
#pragma unroll
  for (int x = 0; x < 3; ++x)  // rot^T
    l[x] = R[x] * d0 + R[3 + x] * d1 + R[6 + x] * d2;
  float nrm = sqrtf(l[0] * l[0] + l[1] * l[1] + l[2] * l[2] + EPSV);
  const int pf = h * PV + p;
  float* crow = cat + (size_t)n * DCAT;
  crow[HCH + 0 * 96 + pf] = l[0];
  crow[HCH + 1 * 96 + pf] = l[1];
  crow[HCH + 2 * 96 + pf] = l[2];
  crow[HCH + 3 * 96 + pf] = nrm;
}

// o_pair[i, h*128+c] = sum_j a[h,i,j] z[i,j,c]   (z pass 2)
// Double-buffered TDM pipeline: tile t+1 streams into LDS while WMMAs eat tile t.
__global__ __launch_bounds__(32)
void k_out_opair(const float* __restrict__ Abuf, const float* __restrict__ z,
                 float* __restrict__ cat) {
  using namespace ipa;
  __shared__ float sa[16 * 32];
  __shared__ float sbz[2][32 * CZ];  // 2 x 16 KB
  const int lane = threadIdx.x;
  const int i = blockIdx.x;
  v8f acc[8];
#pragma unroll
  for (int ct = 0; ct < 8; ++ct) acc[ct] = zero8();

#if IPA_HAS_TDM
  // contiguous 32x128 f32 tile = 4096 x 1
  tdm_load_f32(z + (size_t)i * N * CZ, &sbz[0][0], 32 * CZ, 1, 32 * CZ);
#endif
  for (int t = 0; t < 32; ++t) {
    const int k0 = t * 32;
    for (int e = lane; e < 16 * 32; e += 32) {
      int r = e >> 5, c = e & 31;  // r = head row, pad to 16
      sa[e] = (r < H) ? Abuf[(size_t)r * NN + (size_t)i * N + k0 + c] : 0.f;
    }
#if IPA_HAS_TDM
    if (t + 1 < 32)
      tdm_load_f32(z + ((size_t)i * N + k0 + 32) * CZ, &sbz[(t + 1) & 1][0],
                   32 * CZ, 1, 32 * CZ);
    if (t + 1 < 32) __builtin_amdgcn_s_wait_tensorcnt(1);
    else            __builtin_amdgcn_s_wait_tensorcnt(0);
#else
    const float4* zsrc = (const float4*)(z + ((size_t)i * N + k0) * CZ);
    for (int e = lane; e < (32 * CZ) / 4; e += 32) ((float4*)sbz[t & 1])[e] = zsrc[e];
#endif
    __syncthreads();
    const float* zb = sbz[t & 1];
    v16bf af = frag_a_f32(sa, 32);
#pragma unroll
    for (int ct = 0; ct < 8; ++ct)
      acc[ct] = wmma_bf16(af, frag_b_f32(zb + ct * 16, CZ), acc[ct]);
    __syncthreads();
  }
  const int n = lane & 15, hi = lane >> 4;
#pragma unroll
  for (int ct = 0; ct < 8; ++ct)
#pragma unroll
    for (int v = 0; v < 8; ++v) {
      int m = v + 8 * hi;  // head
      if (m < H)
        cat[(size_t)i * DCAT + HCH + 4 * 96 + m * CZ + ct * 16 + n] = acc[ct][v];
    }
}

// ---------------------------------------------------------------------------
extern "C" void kernel_launch(void* const* d_in, const int* in_sizes, int n_in,
                              void* d_out, int out_size, void* d_ws, size_t ws_size,
                              hipStream_t stream) {
  using namespace ipa;
  (void)in_sizes; (void)n_in; (void)out_size; (void)ws_size;
  const float* s      = (const float*)d_in[0];
  const float* z      = (const float*)d_in[1];
  const float* rot    = (const float*)d_in[2];
  const float* trans  = (const float*)d_in[3];
  const float* mask   = (const float*)d_in[4];
  const float* wq     = (const float*)d_in[5];
  const float* bq     = (const float*)d_in[6];
  const float* wkv    = (const float*)d_in[7];
  const float* bkv    = (const float*)d_in[8];
  const float* wqp    = (const float*)d_in[9];
  const float* bqp    = (const float*)d_in[10];
  const float* wkvp   = (const float*)d_in[11];
  const float* bkvp   = (const float*)d_in[12];
  const float* wb     = (const float*)d_in[13];
  const float* bb     = (const float*)d_in[14];
  const float* head_w = (const float*)d_in[15];
  const float* wout   = (const float*)d_in[16];
  const float* bout   = (const float*)d_in[17];
  float* out = (float*)d_out;

  float* Abuf   = (float*)d_ws;                       // H*N*N
  float* qraw   = Abuf   + (size_t)H * NN;            // N*192
  float* kvraw  = qraw   + (size_t)N * HCH;           // N*384
  float* qpraw  = kvraw  + (size_t)N * H2CH;          // N*144
  float* kvpraw = qpraw  + (size_t)N * HQP3;          // N*432
  float* qn     = kvpraw + (size_t)N * HKVP3;         // N*12
  float* kn     = qn     + (size_t)N * H;             // N*12
  float* vpts   = kn     + (size_t)N * H;             // N*12*32 (padded)
  float* optraw = vpts   + (size_t)N * H * 32;        // N*12*32
  float* cat    = optraw + (size_t)N * H * 32;        // N*2112
  __bf16* qcat  = (__bf16*)(cat + (size_t)N * DCAT);  // H*N*32
  __bf16* kcat  = qcat + (size_t)H * N * 32;          // H*N*32

  const dim3 b32(32);
  k_gemm<<<dim3(N / 64, HCH / 16),   b32, 0, stream>>>(s, CS, wq,   HCH,   bq,   qraw,   HCH,   CS, HCH,   1.f);
  k_gemm<<<dim3(N / 64, H2CH / 16),  b32, 0, stream>>>(s, CS, wkv,  H2CH,  bkv,  kvraw,  H2CH,  CS, H2CH,  1.f);
  k_gemm<<<dim3(N / 64, HQP3 / 16),  b32, 0, stream>>>(s, CS, wqp,  HQP3,  bqp,  qpraw,  HQP3,  CS, HQP3,  1.f);
  k_gemm<<<dim3(N / 64, HKVP3 / 16), b32, 0, stream>>>(s, CS, wkvp, HKVP3, bkvp, kvpraw, HKVP3, CS, HKVP3, 1.f);
  k_bias<<<dim3((unsigned)(NN / 64)), b32, 0, stream>>>(z, wb, bb, Abuf);
  k_points<<<dim3(N), dim3(384), 0, stream>>>(qraw, kvraw, qpraw, kvpraw, rot, trans,
                                              head_w, qn, kn, vpts, qcat, kcat);
  k_logits<<<dim3(N / 64, H), b32, 0, stream>>>(qcat, kcat, qn, kn, mask, head_w, Abuf);
  k_softmax<<<dim3(N, H), dim3(256), 0, stream>>>(Abuf);
  k_out_o<<<dim3(N / 64, H), b32, 0, stream>>>(Abuf, kvraw, cat);
  k_out_opt<<<dim3(N / 64, H), b32, 0, stream>>>(Abuf, vpts, optraw);
  k_opt_finish<<<dim3(N), dim3(96), 0, stream>>>(optraw, rot, trans, cat);
  k_out_opair<<<dim3(N), b32, 0, stream>>>(Abuf, z, cat);
  k_gemm<<<dim3(N / 64, CS / 16), b32, 0, stream>>>(cat, DCAT, wout, CS, bout, out, CS, DCAT, CS, 1.f);
}
